// DynamicDecoder_34840774705401
// MI455X (gfx1250) — compile-verified, matrix operand order checked by reading, not run on hardware
//
#include <hip/hip_runtime.h>
#include <math.h>

typedef __bf16 bf16;
typedef __attribute__((ext_vector_type(8)))  bf16  v8bf;
typedef __attribute__((ext_vector_type(16))) bf16  v16bf;
typedef __attribute__((ext_vector_type(8)))  float v8f;

#define B_SZ   32
#define M_SEQ  600
#define H_DIM  256
#define T_STEPS 4
#define WPB    4   // waves per block for WMMA kernels
#define MT     4   // row tiles per wave in the big fused kernel
#define NT     2   // column groups per wave in the big fused kernel

__device__ __forceinline__ float b2f(bf16 x){ return (float)x; }
__device__ __forceinline__ bf16  f2b(float x){ return (bf16)x; }
__device__ __forceinline__ float sigm(float x){ return 1.0f / (1.0f + __expf(-x)); }

// ---------------------------------------------------------------------------
// 16-lane max reduction via DPP VALU ops (no LDS/ds_bpermute traffic).
// Steps: quad xor1, quad xor2, ROW_HALF_MIRROR (pairs quads within 8),
// ROW_MIRROR (pairs 8-halves within 16).  Never crosses a 16-lane row.
// ---------------------------------------------------------------------------
template<int CTRL>
__device__ __forceinline__ float dpp_max_step(float v){
  int iv = __float_as_int(v);
  int sh = __builtin_amdgcn_update_dpp(iv, iv, CTRL, 0xf, 0xf, true);
  return fmaxf(v, __int_as_float(sh));
}
__device__ __forceinline__ float max_across16(float v){
  v = dpp_max_step<0x0B1>(v);  // quad_perm [1,0,3,2]  (xor 1)
  v = dpp_max_step<0x04E>(v);  // quad_perm [2,3,0,1]  (xor 2)
  v = dpp_max_step<0x141>(v);  // ROW_HALF_MIRROR      (pairs quads in 8)
  v = dpp_max_step<0x140>(v);  // ROW_MIRROR           (pairs 8-halves in 16)
  return v;
}

// ---------------------------------------------------------------------------
// WMMA fragment loader: 16x32 bf16 A/B fragment from a row-major matrix whose
// fragment "rows" run along K with leading dimension ld (elements).
// ISA layout (16-bit A 16x32): lanes 0-15 row M=l, K = kbase+0..7 & +16..23;
// lanes 16-31 row M=l-16, K = kbase+8..15 & +24..31.
// ---------------------------------------------------------------------------
__device__ __forceinline__ v16bf load_frag_bf16(const bf16* X, int ld, int row0, int kbase){
  int l  = threadIdx.x & 31;
  int r  = row0 + (l & 15);
  int kb = kbase + ((l >> 4) << 3);
  const v8bf* p0 = (const v8bf*)(X + (size_t)r * ld + kb);
  const v8bf* p1 = (const v8bf*)(X + (size_t)r * ld + kb + 16);
  v8bf lo = *p0, hi = *p1;
  v16bf f;
#pragma unroll
  for (int i = 0; i < 8; ++i){ f[i] = lo[i]; f[i + 8] = hi[i]; }
  return f;
}

// ---------------------------------------------------------------------------
// Generic GEMM: C[M,N] = A[M,K] @ W[N,K]^T (+bias) (+=existing) (tanh)
// 4 waves / 128-thread block, one 16x16 D tile per wave (small GEMMs only).
// ---------------------------------------------------------------------------
__global__ void gemm_bf16_kernel(const bf16* __restrict__ A, int lda,
                                 const bf16* __restrict__ W, int ldw,
                                 float* __restrict__ Cf, bf16* __restrict__ Cb, int ldc,
                                 int N, int K,
                                 const float* __restrict__ bias,
                                 int accumulate, int act_tanh){
  int wave  = threadIdx.x >> 5;
  int row0  = blockIdx.x * 16;
  int col0  = (blockIdx.y * WPB + wave) * 16;
  v8f acc = {};
  for (int k = 0; k < K; k += 32){
    v16bf a = load_frag_bf16(A, lda, row0, k);
    v16bf b = load_frag_bf16(W, ldw, col0, k);
    acc = __builtin_amdgcn_wmma_f32_16x16x32_bf16(false, a, false, b, (short)0, acc, false, false);
  }
  int l  = threadIdx.x & 31;
  int n  = col0 + (l & 15);
  int mb = row0 + ((l >> 4) << 3);
  float bv = bias ? bias[n] : 0.0f;
#pragma unroll
  for (int r = 0; r < 8; ++r){
    size_t idx = (size_t)(mb + r) * ldc + n;
    float v = acc[r] + bv;
    if (accumulate) v += Cf[idx];
    if (act_tanh)   v = tanhf(v);
    if (Cf) Cf[idx] = v;
    if (Cb) Cb[idx] = f2b(v);
  }
}

// ---------------------------------------------------------------------------
// Fused GEMM + (bias | per-batch vector) + maxpool over groups of 16 columns.
// Register-blocked: each wave owns MT x NT 16x16 D tiles (8 accumulators),
// so each K-step does MT+NT fragment loads feeding MT*NT WMMAs (1.5 b128
// loads per WMMA instead of 4).  A 16x16 WMMA tile == one maxout group, so
// the pool is a 4-step DPP max on the accumulator fragment.
// ---------------------------------------------------------------------------
__global__ void gemm_maxpool16_kernel(const bf16* __restrict__ A, int lda,
                                      const bf16* __restrict__ W, int ldw,
                                      bf16* __restrict__ out,
                                      int N, int K,
                                      const float* __restrict__ bias,
                                      const float* __restrict__ perbatch){
  int wave  = threadIdx.x >> 5;
  int row0  = blockIdx.x * (16 * MT);
  int jbase = (blockIdx.y * WPB + wave) * NT;
  v8f acc[MT][NT] = {};
  for (int k = 0; k < K; k += 32){
    v16bf af[MT], bfr[NT];
#pragma unroll
    for (int i = 0; i < MT; ++i)  af[i]  = load_frag_bf16(A, lda, row0 + i * 16, k);
#pragma unroll
    for (int jj = 0; jj < NT; ++jj) bfr[jj] = load_frag_bf16(W, ldw, (jbase + jj) * 16, k);
#pragma unroll
    for (int i = 0; i < MT; ++i)
#pragma unroll
      for (int jj = 0; jj < NT; ++jj)
        acc[i][jj] = __builtin_amdgcn_wmma_f32_16x16x32_bf16(
            false, af[i], false, bfr[jj], (short)0, acc[i][jj], false, false);
  }
  int l  = threadIdx.x & 31;
  int mb = (l >> 4) << 3;
  int No = N >> 4;
  // A 64-row supertile crosses at most one batch boundary: avoid per-element
  // integer division for the per-batch add vector.
  int b0  = row0 / M_SEQ;
  int rem = row0 - b0 * M_SEQ;
#pragma unroll
  for (int jj = 0; jj < NT; ++jj){
    int j = jbase + jj;
    int n = j * 16 + (l & 15);
    float bv = bias ? bias[n] : 0.0f;
    float add0 = perbatch ? perbatch[(size_t)b0 * N + n] : 0.0f;
    float add1 = perbatch ? perbatch[(size_t)(b0 + 1 < B_SZ ? b0 + 1 : b0) * N + n] : 0.0f;
#pragma unroll
    for (int i = 0; i < MT; ++i){
      float vals[8];
#pragma unroll
      for (int r = 0; r < 8; ++r){
        int d = i * 16 + mb + r;
        float add = (rem + d >= M_SEQ) ? add1 : add0;
        vals[r] = acc[i][jj][r] + bv + add;
      }
#pragma unroll
      for (int r = 0; r < 8; ++r)
        vals[r] = max_across16(vals[r]);
      if ((l & 15) == 0){
#pragma unroll
        for (int r = 0; r < 8; ++r)
          out[(size_t)(row0 + i * 16 + mb + r) * No + j] = f2b(vals[r]);
      }
    }
  }
}

// ---------------------------------------------------------------------------
// alpha[row] = max_k( [m1row | m2row] . W12[k,:] + b12[k] ) + mask
// One wave per row; lanes (k = l&15, half = l>>4) split the 512-dot in two.
// ---------------------------------------------------------------------------
__global__ void alpha_kernel(const bf16* __restrict__ m1, const bf16* __restrict__ m2,
                             const bf16* __restrict__ W12, const float* __restrict__ b12,
                             const int* __restrict__ dmask, float* __restrict__ alpha){
  int wave = threadIdx.x >> 5;
  int row  = blockIdx.x * WPB + wave;
  int l = threadIdx.x & 31;
  int k = l & 15, half = l >> 4;
  const bf16* cat = (half ? m2 : m1) + (size_t)row * H_DIM;
  const bf16* w   = W12 + (size_t)k * (2 * H_DIM) + half * H_DIM;
  float s = 0.f;
  for (int c = 0; c < H_DIM; ++c) s += b2f(cat[c]) * b2f(w[c]);
  s += __shfl_xor(s, 16, 32);   // combine the two K-halves (crosses DPP row)
  s += b12[k];
  s = max_across16(s);
  if (l == 0){
    int b = row / M_SEQ, m = row % M_SEQ;
    alpha[row] = s + (dmask[b * M_SEQ + m] ? 0.f : -1e30f);
  }
}

// ---------------------------------------------------------------------------
// Per-batch: log-softmax target value, argmax (first-index tie-break),
// mask/index state update, gather U[b, idx] into u_cat / hcat slots.
// ---------------------------------------------------------------------------
__global__ void select_kernel(const float* __restrict__ alpha, const float* __restrict__ U,
                              const int* __restrict__ span,
                              int* __restrict__ idx_state, int* __restrict__ mask_state,
                              int* __restrict__ res_idx, int* __restrict__ res_mask,
                              float* __restrict__ negls,
                              bf16* __restrict__ ucat_b, bf16* __restrict__ hcat_b,
                              int t, int branch){
  int b = blockIdx.x;
  int tid = threadIdx.x;
  __shared__ float sv[256];
  __shared__ int   si[256];
  // pass 1: max + first-arg index
  float best = -INFINITY; int bidx = 0x7fffffff;
  for (int m = tid; m < M_SEQ; m += 256){
    float v = alpha[b * M_SEQ + m];
    if (v > best || (v == best && m < bidx)){ best = v; bidx = m; }
  }
  sv[tid] = best; si[tid] = bidx; __syncthreads();
  for (int s = 128; s > 0; s >>= 1){
    if (tid < s){
      if (sv[tid + s] > sv[tid] || (sv[tid + s] == sv[tid] && si[tid + s] < si[tid])){
        sv[tid] = sv[tid + s]; si[tid] = si[tid + s];
      }
    }
    __syncthreads();
  }
  float maxv = sv[0]; int amax = si[0]; __syncthreads();
  // pass 2: sum of exp
  float se = 0.f;
  for (int m = tid; m < M_SEQ; m += 256) se += __expf(alpha[b * M_SEQ + m] - maxv);
  sv[tid] = se; __syncthreads();
  for (int s = 128; s > 0; s >>= 1){
    if (tid < s) sv[tid] += sv[tid + s];
    __syncthreads();
  }
  float lse = maxv + __logf(sv[0]);
  __shared__ int s_idx;
  if (tid == 0){
    int tgt = span[b * 2 + branch];
    negls[b] = -(alpha[b * M_SEQ + tgt] - lse);
    int idx = amax, mnew;
    if (t == 0){ mnew = 1; }
    else {
      int mold = mask_state[b];
      idx = idx * mold;
      int prev = idx_state[b] * mold;
      mnew = (idx != prev) ? 1 : 0;
    }
    mask_state[b] = mnew;
    idx_state[b]  = idx;
    res_idx[t * B_SZ + b]  = idx;
    res_mask[t * B_SZ + b] = mnew;
    s_idx = idx;
  }
  __syncthreads();
  int idx = s_idx;
  const float* urow = U + ((size_t)b * M_SEQ + idx) * 512;
  for (int q = tid; q < 512; q += 256){
    bf16 v = f2b(urow[q]);
    ucat_b[b * 1024 + branch * 512 + q] = v;
    hcat_b[b * 1280 + 256 + branch * 512 + q] = v;
  }
}

__global__ void loss_accum_kernel(const float* __restrict__ negls,
                                  const int* __restrict__ mask_state,
                                  float* __restrict__ loss_vec){
  int b = threadIdx.x;
  float s = negls[b];
  for (int o = 16; o > 0; o >>= 1) s += __shfl_xor(s, o, 32);
  loss_vec[b] += (s / 32.0f) * (float)mask_state[b];
}

__global__ void lstm_act_kernel(const float* __restrict__ g,
                                const float* __restrict__ b_ih, const float* __restrict__ b_hh,
                                float* __restrict__ c, float* __restrict__ h,
                                bf16* __restrict__ h_bf, bf16* __restrict__ hcat_b){
  int b = blockIdx.x, j = threadIdx.x;
  const float* gb = g + (size_t)b * 1024;
  float gi = gb[j]       + b_ih[j]       + b_hh[j];
  float gf = gb[256 + j] + b_ih[256 + j] + b_hh[256 + j];
  float gg = gb[512 + j] + b_ih[512 + j] + b_hh[512 + j];
  float go = gb[768 + j] + b_ih[768 + j] + b_hh[768 + j];
  float cn = sigm(gf) * c[b * 256 + j] + sigm(gi) * tanhf(gg);
  float hn = sigm(go) * tanhf(cn);
  c[b * 256 + j] = cn;
  h[b * 256 + j] = hn;
  h_bf[b * 256 + j] = f2b(hn);
  hcat_b[b * 1280 + j] = f2b(hn);
}

__global__ void init_kernel(const float* __restrict__ U, const int* __restrict__ dmask,
                            int* __restrict__ s_i, int* __restrict__ e_i,
                            int* __restrict__ mask_s, int* __restrict__ mask_e,
                            float* __restrict__ h, float* __restrict__ c,
                            bf16* __restrict__ h_bf, bf16* __restrict__ hcat_b,
                            bf16* __restrict__ ucat_b, float* __restrict__ loss_vec){
  int b = blockIdx.x, tid = threadIdx.x;
  __shared__ int ssum[256];
  int s = 0;
  for (int m = tid; m < M_SEQ; m += 256) s += dmask[b * M_SEQ + m];
  ssum[tid] = s; __syncthreads();
  for (int st = 128; st > 0; st >>= 1){
    if (tid < st) ssum[tid] += ssum[tid + st];
    __syncthreads();
  }
  __shared__ int se0;
  if (tid == 0){
    int e0 = ssum[0] - 1;
    e_i[b] = e0; s_i[b] = 0; mask_s[b] = 1; mask_e[b] = 1; loss_vec[b] = 0.f;
    se0 = e0;
  }
  __syncthreads();
  int e0 = se0;
  h[b * 256 + tid] = 0.f;
  c[b * 256 + tid] = 0.f;
  h_bf[b * 256 + tid] = f2b(0.f);
  hcat_b[b * 1280 + tid] = f2b(0.f);
  for (int q = tid; q < 512; q += 256){
    bf16 vs = f2b(U[((size_t)b * M_SEQ + 0) * 512 + q]);
    bf16 ve = f2b(U[((size_t)b * M_SEQ + e0) * 512 + q]);
    ucat_b[b * 1024 + q] = vs;
    ucat_b[b * 1024 + 512 + q] = ve;
    hcat_b[b * 1280 + 256 + q] = vs;
    hcat_b[b * 1280 + 768 + q] = ve;
  }
}

__global__ void finalize_kernel(const int* __restrict__ rs_i, const int* __restrict__ rs_m,
                                const int* __restrict__ re_i, const int* __restrict__ re_m,
                                const float* __restrict__ loss_vec, float* __restrict__ out){
  int b = threadIdx.x;
  int ps = 0, pe = 0;
  for (int t = 0; t < T_STEPS; ++t){ ps += rs_m[t * B_SZ + b]; pe += re_m[t * B_SZ + b]; }
  int p1 = rs_i[(ps - 1) * B_SZ + b];
  int p2 = re_i[(pe - 1) * B_SZ + b];
  float lv = loss_vec[b] * (1.0f / T_STEPS);
  float s = lv;
  for (int o = 16; o > 0; o >>= 1) s += __shfl_xor(s, o, 32);
  if (b == 0) out[0] = s / (float)B_SZ;
  out[1 + b] = (float)p1;
  out[1 + B_SZ + b] = (float)p2;
}

__global__ void cvt_kernel(const float* __restrict__ src, bf16* __restrict__ dst, int n){
  int i = blockIdx.x * 256 + threadIdx.x;
  if (i < n) dst[i] = f2b(src[i]);
}

// ---------------------------------------------------------------------------
extern "C" void kernel_launch(void* const* d_in, const int* in_sizes, int n_in,
                              void* d_out, int out_size, void* d_ws, size_t ws_size,
                              hipStream_t stream){
  const float* U      = (const float*)d_in[0];
  const int*   dmask  = (const int*)  d_in[1];
  const int*   span   = (const int*)  d_in[2];
  const float* W_ih   = (const float*)d_in[3];
  const float* W_hh   = (const float*)d_in[4];
  const float* b_ih   = (const float*)d_in[5];
  const float* b_hh   = (const float*)d_in[6];
  const float* Wlin_f[2]  = { (const float*)d_in[7],  (const float*)d_in[14] };
  const float* Wm1_f[2]   = { (const float*)d_in[8],  (const float*)d_in[15] };
  const float* bm1_f[2]   = { (const float*)d_in[9],  (const float*)d_in[16] };
  const float* Wm2_f[2]   = { (const float*)d_in[10], (const float*)d_in[17] };
  const float* bm2_f[2]   = { (const float*)d_in[11], (const float*)d_in[18] };
  const float* Wm12_f[2]  = { (const float*)d_in[12], (const float*)d_in[19] };
  const float* bm12_f[2]  = { (const float*)d_in[13], (const float*)d_in[20] };

  char* ws = (char*)d_ws;
  size_t off = 0;
  auto alloc = [&](size_t bytes)->char*{
    char* p = ws + off;
    off = (off + bytes + 255) & ~(size_t)255;
    return p;
  };

  const size_t ROWS = (size_t)B_SZ * M_SEQ;           // 19200
  bf16* U_b     = (bf16*)alloc(ROWS * 512 * 2);
  bf16* Wih_b   = (bf16*)alloc(1024 * 1024 * 2);
  bf16* Whh_b   = (bf16*)alloc(1024 * 256 * 2);
  bf16* Wlin_b[2], *Wm1_b[2], *Wm2_b[2], *Wm12_b[2];
  for (int s = 0; s < 2; ++s){
    Wlin_b[s] = (bf16*)alloc(256 * 1280 * 2);
    Wm1_b[s]  = (bf16*)alloc((size_t)4096 * 768 * 2);
    Wm2_b[s]  = (bf16*)alloc((size_t)4096 * 256 * 2);
    Wm12_b[s] = (bf16*)alloc(16 * 512 * 2);
  }
  float* g      = (float*)alloc(B_SZ * 1024 * 4);
  float* h_f    = (float*)alloc(B_SZ * 256 * 4);
  float* c_f    = (float*)alloc(B_SZ * 256 * 4);
  bf16*  h_b    = (bf16*) alloc(B_SZ * 256 * 2);
  bf16*  ucat_b = (bf16*) alloc(B_SZ * 1024 * 2);
  bf16*  hcat_b = (bf16*) alloc(B_SZ * 1280 * 2);
  bf16*  r_b    = (bf16*) alloc(B_SZ * 256 * 2);
  float* rW1    = (float*)alloc(B_SZ * 4096 * 4);
  bf16*  m1_b   = (bf16*) alloc(ROWS * 256 * 2);
  bf16*  m2_b   = (bf16*) alloc(ROWS * 256 * 2);
  float* alpha  = (float*)alloc(ROWS * 4);
  int*   s_i    = (int*)  alloc(B_SZ * 4);
  int*   e_i    = (int*)  alloc(B_SZ * 4);
  int*   mask_s = (int*)  alloc(B_SZ * 4);
  int*   mask_e = (int*)  alloc(B_SZ * 4);
  int*   res_si = (int*)  alloc(T_STEPS * B_SZ * 4);
  int*   res_sm = (int*)  alloc(T_STEPS * B_SZ * 4);
  int*   res_ei = (int*)  alloc(T_STEPS * B_SZ * 4);
  int*   res_em = (int*)  alloc(T_STEPS * B_SZ * 4);
  float* negls  = (float*)alloc(B_SZ * 4);
  float* lossv  = (float*)alloc(B_SZ * 4);

  auto cvt = [&](const float* src, bf16* dst, size_t n){
    cvt_kernel<<<dim3((unsigned)((n + 255) / 256)), 256, 0, stream>>>(src, dst, (int)n);
  };
  cvt(U, U_b, ROWS * 512);
  cvt(W_ih, Wih_b, 1024 * 1024);
  cvt(W_hh, Whh_b, 1024 * 256);
  for (int s = 0; s < 2; ++s){
    cvt(Wlin_f[s], Wlin_b[s], 256 * 1280);
    cvt(Wm1_f[s],  Wm1_b[s],  (size_t)4096 * 768);
    cvt(Wm2_f[s],  Wm2_b[s],  (size_t)4096 * 256);
    cvt(Wm12_f[s], Wm12_b[s], 16 * 512);
  }

  init_kernel<<<B_SZ, 256, 0, stream>>>(U, dmask, s_i, e_i, mask_s, mask_e,
                                        h_f, c_f, h_b, hcat_b, ucat_b, lossv);

  // Big fused kernels: 64-row supertiles x (NT*WPB) column groups per block.
  const dim3 grid_big((unsigned)(ROWS / (16 * MT)), 4096 / (16 * NT * WPB));  // (300, 32)

  for (int t = 0; t < T_STEPS; ++t){
    // LSTM: g = u_cat @ W_ih^T ; g += h @ W_hh^T ; activations
    gemm_bf16_kernel<<<dim3(2, 16), 128, 0, stream>>>(ucat_b, 1024, Wih_b, 1024,
        g, nullptr, 1024, 1024, 1024, nullptr, 0, 0);
    gemm_bf16_kernel<<<dim3(2, 16), 128, 0, stream>>>(h_b, 256, Whh_b, 256,
        g, nullptr, 1024, 1024, 256, nullptr, 1, 0);
    lstm_act_kernel<<<B_SZ, 256, 0, stream>>>(g, b_ih, b_hh, c_f, h_f, h_b, hcat_b);

    for (int br = 0; br < 2; ++br){
      // r = tanh(hcat @ Wlin^T)  (B x 256)
      gemm_bf16_kernel<<<dim3(2, 4), 128, 0, stream>>>(hcat_b, 1280, Wlin_b[br], 1280,
          nullptr, r_b, 256, 256, 1280, nullptr, 0, 1);
      // rW1 = r @ Wm1[:,512:768]^T + b_m1  (B x 4096)
      gemm_bf16_kernel<<<dim3(2, 64), 128, 0, stream>>>(r_b, 256, Wm1_b[br] + 512, 768,
          rW1, nullptr, 4096, 4096, 256, bm1_f[br], 0, 0);
      // m1 = maxpool16( U @ Wm1[:,:512]^T + rW1 )   (ROWS x 256)
      gemm_maxpool16_kernel<<<grid_big, 128, 0, stream>>>(U_b, 512, Wm1_b[br], 768,
          m1_b, 4096, 512, nullptr, rW1);
      // m2 = maxpool16( m1 @ Wm2^T + b_m2 )         (ROWS x 256)
      gemm_maxpool16_kernel<<<grid_big, 128, 0, stream>>>(m1_b, 256, Wm2_b[br], 256,
          m2_b, 4096, 256, bm2_f[br], nullptr);
      // alpha = max_k([m1|m2] @ Wm12^T + b_m12) + mask
      alpha_kernel<<<dim3((unsigned)(ROWS / WPB)), 128, 0, stream>>>(m1_b, m2_b,
          Wm12_b[br], bm12_f[br], dmask, alpha);
      // log-softmax + argmax + state update + gather
      select_kernel<<<B_SZ, 256, 0, stream>>>(alpha, U, span,
          br ? e_i : s_i, br ? mask_e : mask_s,
          br ? res_ei : res_si, br ? res_em : res_sm,
          negls, ucat_b, hcat_b, t, br);
      loss_accum_kernel<<<1, 32, 0, stream>>>(negls, br ? mask_e : mask_s, lossv);
    }
  }

  finalize_kernel<<<1, 32, 0, stream>>>(res_si, res_sm, res_ei, res_em, lossv, (float*)d_out);
}
